// Token_transformer_17763984736762
// MI455X (gfx1250) — matrix operand order repaired and checked by reading, not compiled, not run on hardware
//
#include <hip/hip_runtime.h>
#include <hip/hip_bf16.h>
#include <math.h>

// Problem constants (from reference)
#define BATCH 8
#define NTOK  3136          // = 196 * 16 = 98 * 32
#define DIM   147
#define IND   64
#define QTILES_PER_B 196    // NTOK / 16
#define EPS   1e-5f

typedef __attribute__((ext_vector_type(16))) _Float16 v16h;
typedef __attribute__((ext_vector_type(8)))  _Float16 v8h;
typedef __attribute__((ext_vector_type(8)))  float    v8f;

union V16U { v16h v; v8h h[2]; };

__device__ __forceinline__ float wave_sum32(float x) {
    #pragma unroll
    for (int m = 16; m >= 1; m >>= 1) x += __shfl_xor(x, m, 32);
    return x;
}

// raw v_exp_f32 (no libm subnormal fixup -- flush-to-zero is what softmax wants)
__device__ __forceinline__ float fast_exp2(float x) {
    return __builtin_amdgcn_exp2f(x);
}

__device__ __forceinline__ v8f wmma_f16(v16h a, v16h b, v8f c) {
    // v_wmma_f32_16x16x32_f16  (neg_a, A, neg_b, B, c_mod, C, reuse_a, reuse_b)
    return __builtin_amdgcn_wmma_f32_16x16x32_f16(false, a, false, b, (short)0, c, false, false);
}

// ---------------------------------------------------------------------------
// Kernel 1: LayerNorm(147) + QKV projection. One wave per token row.
//   qT -> d-major  f16 [B][64][N], PRE-SCALED by scale*log2(e) (B-op of S^T-WMMA)
//   k  -> row-major f16 [B][N][64]                             (A-op of S^T-WMMA)
//   vT -> d-major  f16 [B][64][N]                              (A-op of O^T-WMMA)
//   v32-> row-major f32 [B][N][64]  (residual)
// ---------------------------------------------------------------------------
__global__ __launch_bounds__(256) void ln_qkv_kernel(
    const float* __restrict__ x,
    const float* __restrict__ n1w, const float* __restrict__ n1b,
    const float* __restrict__ qkvw, const float* __restrict__ scale_p,
    _Float16* __restrict__ qT, _Float16* __restrict__ kf,
    _Float16* __restrict__ vT, float* __restrict__ v32)
{
    __shared__ float hrow[8][160];
    const int wave = threadIdx.x >> 5, lane = threadIdx.x & 31;
    const int row  = blockIdx.x * 8 + wave;            // 0 .. B*N-1
    const int b = row / NTOK, n = row % NTOK;
    const float qscale = scale_p[0] * 1.44269504088896f;   // fold scale & log2e into Q
    const float* xr = x + (size_t)row * DIM;

    float s = 0.f;
    for (int k = lane; k < DIM; k += 32) s += xr[k];
    const float mean = wave_sum32(s) * (1.0f / DIM);
    float v2 = 0.f;
    for (int k = lane; k < DIM; k += 32) { float d = xr[k] - mean; v2 = fmaf(d, d, v2); }
    const float rstd = rsqrtf(wave_sum32(v2) * (1.0f / DIM) + EPS);
    for (int k = lane; k < DIM; k += 32)
        hrow[wave][k] = (xr[k] - mean) * rstd * n1w[k] + n1b[k];
    // same-wave LDS ops are in-order: no barrier needed

    const float* hr = hrow[wave];
    #pragma unroll
    for (int j = 0; j < 6; ++j) {
        const int o = lane + 32 * j;                   // 0..191
        const float* wrow = qkvw + (size_t)o * DIM;
        float acc = 0.f;
        for (int k = 0; k < DIM; ++k) acc = fmaf(hr[k], wrow[k], acc);
        if (o < 64) {
            qT[(size_t)(b * 64 + o) * NTOK + n] = (_Float16)(acc * qscale);
        } else if (o < 128) {
            kf[((size_t)row << 6) + (o - 64)] = (_Float16)acc;
        } else {
            const int d = o - 128;
            vT [(size_t)(b * 64 + d) * NTOK + n] = (_Float16)acc;
            v32[((size_t)row << 6) + d] = acc;
        }
    }
}

// ---------------------------------------------------------------------------
// One 32-key flash-attention step, TRANSPOSED: S^T = K * Q^T, O^T += V^T * P^T.
// In C-layout of S^T, each lane owns one query column -> softmax reductions are
// in-register (plus a single xor-16 partner exchange); rescale is per-lane.
// MASKED variant (diagonal exclusion) instantiated for one tile only.
// ---------------------------------------------------------------------------
template <bool MASKED>
__device__ __forceinline__ void attn_step(
    int jb, int qglob, int n16, int hi,
    const _Float16* __restrict__ kb, const _Float16* __restrict__ vTb,
    _Float16* __restrict__ ptile,
    v16h Qb0, v16h Qb1,
    v8f& o0, v8f& o1, v8f& o2, v8f& o3,
    float& mrow, float& lrow)
{
    const int lane = hi * 16 + n16;

    // --- K as A-operand: two 16-key tiles x two 32-d chunks (row-major k) ---
    const _Float16* kr0 = kb + (((size_t)(jb + n16))      << 6);
    const _Float16* kr1 = kb + (((size_t)(jb + 16 + n16)) << 6);
    V16U Ka00, Ka01, Ka10, Ka11;
    Ka00.h[0] = *(const v8h*)(kr0 +      8 * hi); Ka00.h[1] = *(const v8h*)(kr0 + 16 + 8 * hi);
    Ka01.h[0] = *(const v8h*)(kr0 + 32 + 8 * hi); Ka01.h[1] = *(const v8h*)(kr0 + 48 + 8 * hi);
    Ka10.h[0] = *(const v8h*)(kr1 +      8 * hi); Ka10.h[1] = *(const v8h*)(kr1 + 16 + 8 * hi);
    Ka11.h[0] = *(const v8h*)(kr1 + 32 + 8 * hi); Ka11.h[1] = *(const v8h*)(kr1 + 48 + 8 * hi);
    __builtin_prefetch(kr0 + 2048, 0, 0);   // next 32-key tile (speculative-safe)
    __builtin_prefetch(kr1 + 2048, 0, 0);

    // --- S^T tiles: rows = keys (r + 8*hi within tile), col = query (lane n16) ---
    v8f st0 = {}, st1 = {};
    st0 = wmma_f16(Ka00.v, Qb0, st0);
    st0 = wmma_f16(Ka01.v, Qb1, st0);
    st1 = wmma_f16(Ka10.v, Qb0, st1);
    st1 = wmma_f16(Ka11.v, Qb1, st1);

    if (MASKED) {
        #pragma unroll
        for (int r = 0; r < 8; ++r) {
            st0[r] = (jb +      r + 8 * hi == qglob) ? -1e30f : st0[r];
            st1[r] = (jb + 16 + r + 8 * hi == qglob) ? -1e30f : st1[r];
        }
    }

    // --- per-query online softmax: in-register + one xor-16 partner merge ---
    float t = fmaxf(st0[0], st1[0]);
    #pragma unroll
    for (int r = 1; r < 8; ++r) t = fmaxf(t, fmaxf(st0[r], st1[r]));
    t = fmaxf(t, __shfl_xor(t, 16, 32));
    const float mnew = fmaxf(mrow, t);
    const float c = fast_exp2(mrow - mnew);      // base-2: Q pre-scaled by log2e
    float p0[8], p1[8], rs = 0.f;
    #pragma unroll
    for (int r = 0; r < 8; ++r) {
        p0[r] = fast_exp2(st0[r] - mnew);
        p1[r] = fast_exp2(st1[r] - mnew);
        rs += p0[r] + p1[r];
    }
    rs += __shfl_xor(rs, 16, 32);
    lrow = lrow * c + rs;
    mrow = mnew;
    o0 *= c; o1 *= c; o2 *= c; o3 *= c;          // c is uniform per lane

    // --- P^T -> LDS (32 j-rows x 16 q-cols), re-read as B-operand ---
    #pragma unroll
    for (int r = 0; r < 8; ++r) {
        ptile[(r + 8 * hi) * 16 + n16]        = (_Float16)p0[r];
        ptile[(16 + r + 8 * hi) * 16 + n16]   = (_Float16)p1[r];
    }
    v16h Pb;   // lane n16 = q col, element s = j in table order
    #pragma unroll
    for (int s = 0; s < 16; ++s)
        Pb[s] = ptile[((s < 8) ? (8 * hi + s) : (16 + 8 * hi + (s - 8))) * 16 + n16];

    // --- O^T += V^T * P^T : A-operand from d-major vT, 4 d-tiles ---
    const _Float16* vr0 = vTb + (size_t)(n16)      * NTOK + jb;
    const _Float16* vr1 = vTb + (size_t)(16 + n16) * NTOK + jb;
    const _Float16* vr2 = vTb + (size_t)(32 + n16) * NTOK + jb;
    const _Float16* vr3 = vTb + (size_t)(48 + n16) * NTOK + jb;
    V16U Va;
    Va.h[0] = *(const v8h*)(vr0 + 8 * hi); Va.h[1] = *(const v8h*)(vr0 + 16 + 8 * hi);
    o0 = wmma_f16(Va.v, Pb, o0);
    Va.h[0] = *(const v8h*)(vr1 + 8 * hi); Va.h[1] = *(const v8h*)(vr1 + 16 + 8 * hi);
    o1 = wmma_f16(Va.v, Pb, o1);
    Va.h[0] = *(const v8h*)(vr2 + 8 * hi); Va.h[1] = *(const v8h*)(vr2 + 16 + 8 * hi);
    o2 = wmma_f16(Va.v, Pb, o2);
    Va.h[0] = *(const v8h*)(vr3 + 8 * hi); Va.h[1] = *(const v8h*)(vr3 + 16 + 8 * hi);
    o3 = wmma_f16(Va.v, Pb, o3);
    __builtin_prefetch(vr0 + 32, 0, 0);
    __builtin_prefetch(vr2 + 32, 0, 0);
}

// ---------------------------------------------------------------------------
// Kernel 2: transposed flash attention. One wave per 16-query tile.
// Q^T B-operand loaded ONCE per wave (outside the key loop).
// ---------------------------------------------------------------------------
__global__ __launch_bounds__(128) void flash_attn_kernel(
    const _Float16* __restrict__ qT, const _Float16* __restrict__ kf,
    const _Float16* __restrict__ vT, float* __restrict__ ow)
{
    __shared__ __align__(32) _Float16 pshm[4][32 * 16];   // per-wave P^T staging
    const int wave = threadIdx.x >> 5, lane = threadIdx.x & 31;
    const int wgid  = blockIdx.x * 4 + wave;              // 0 .. 8*196-1
    const int b     = wgid / QTILES_PER_B;
    const int qbase = (wgid % QTILES_PER_B) * 16;
    const int n16 = lane & 15, hi = lane >> 4;
    const int qglob = qbase + n16;                        // this lane's query

    // Q^T as B-operand: lane n16 = query col, element s = d in table order; two 32-d chunks
    const _Float16* qTb = qT + (size_t)b * 64 * NTOK;
    v16h Qb0, Qb1;
    #pragma unroll
    for (int s = 0; s < 16; ++s) {
        const int d = (s < 8) ? (8 * hi + s) : (16 + 8 * hi + (s - 8));
        Qb0[s] = qTb[(size_t)d        * NTOK + qglob];
        Qb1[s] = qTb[(size_t)(32 + d) * NTOK + qglob];
    }

    v8f o0 = {}, o1 = {}, o2 = {}, o3 = {};
    float mrow = -1e30f, lrow = 0.f;

    const _Float16* kb  = kf + (((size_t)b * NTOK) << 6);
    const _Float16* vTb = vT + (size_t)b * 64 * NTOK;
    _Float16* ptile = pshm[wave];

    const int diagTile = qbase & ~31;   // the single 32-key tile crossing the diagonal
    for (int jb = 0; jb < diagTile; jb += 32)
        attn_step<false>(jb, qglob, n16, hi, kb, vTb, ptile, Qb0, Qb1,
                         o0, o1, o2, o3, mrow, lrow);
    attn_step<true>(diagTile, qglob, n16, hi, kb, vTb, ptile, Qb0, Qb1,
                    o0, o1, o2, o3, mrow, lrow);
    for (int jb = diagTile + 32; jb < NTOK; jb += 32)
        attn_step<false>(jb, qglob, n16, hi, kb, vTb, ptile, Qb0, Qb1,
                         o0, o1, o2, o3, mrow, lrow);

    // O^T C-layout: lane = query col, rows = d = dtile*16 + r + 8*hi.
    // For a fixed lane the 8 r's are CONTIGUOUS d's in row-major O[n][64].
    const float inv = 1.0f / lrow;
    float* orow = ow + ((size_t)(b * NTOK + qglob) << 6);
    #pragma unroll
    for (int r = 0; r < 8; ++r) {
        orow[     r + 8 * hi] = o0[r] * inv;
        orow[16 + r + 8 * hi] = o1[r] * inv;
        orow[32 + r + 8 * hi] = o2[r] * inv;
        orow[48 + r + 8 * hi] = o3[r] * inv;
    }
}

// ---------------------------------------------------------------------------
// Kernel 3: proj + residual(v), LN(64), fc1 + exact GELU, fc2 + residual.
// One wave per row; each lane owns output dims {lane, lane+32}.
// ---------------------------------------------------------------------------
__global__ __launch_bounds__(256) void mlp_kernel(
    const float* __restrict__ v32, const float* __restrict__ ow,
    const float* __restrict__ projw, const float* __restrict__ projb,
    const float* __restrict__ n2w,  const float* __restrict__ n2b,
    const float* __restrict__ fc1w, const float* __restrict__ fc1b,
    const float* __restrict__ fc2w, const float* __restrict__ fc2b,
    float* __restrict__ out)
{
    __shared__ float Wp[64 * 64], W1[64 * 64], W2[64 * 64];
    __shared__ float rb1[8][64], rb2[8][64];
    for (int i = threadIdx.x; i < 4096; i += 256) {
        Wp[i] = projw[i]; W1[i] = fc1w[i]; W2[i] = fc2w[i];
    }
    __syncthreads();

    const int wave = threadIdx.x >> 5, lane = threadIdx.x & 31;
    const size_t row = (size_t)blockIdx.x * 8 + wave;
    const int d0 = lane, d1 = lane + 32;
    const float* orow = ow  + (row << 6);
    const float* vrow = v32 + (row << 6);

    rb1[wave][d0] = orow[d0]; rb1[wave][d1] = orow[d1];
    float a0 = projb[d0], a1 = projb[d1];
    #pragma unroll 8
    for (int k = 0; k < 64; ++k) {
        const float h = rb1[wave][k];
        a0 = fmaf(h, Wp[d0 * 64 + k], a0);
        a1 = fmaf(h, Wp[d1 * 64 + k], a1);
    }
    const float xa0 = vrow[d0] + a0;       // residual uses v (num_heads==1)
    const float xa1 = vrow[d1] + a1;

    const float mu = wave_sum32(xa0 + xa1) * (1.0f / 64.0f);
    const float e0 = xa0 - mu, e1 = xa1 - mu;
    const float var = wave_sum32(e0 * e0 + e1 * e1) * (1.0f / 64.0f);
    const float rstd = rsqrtf(var + EPS);
    rb2[wave][d0] = e0 * rstd * n2w[d0] + n2b[d0];
    rb2[wave][d1] = e1 * rstd * n2w[d1] + n2b[d1];

    float g0 = fc1b[d0], g1 = fc1b[d1];
    #pragma unroll 8
    for (int k = 0; k < 64; ++k) {
        const float h = rb2[wave][k];
        g0 = fmaf(h, W1[d0 * 64 + k], g0);
        g1 = fmaf(h, W1[d1 * 64 + k], g1);
    }
    g0 = 0.5f * g0 * (1.0f + erff(g0 * 0.70710678118654752f));   // exact GELU
    g1 = 0.5f * g1 * (1.0f + erff(g1 * 0.70710678118654752f));
    rb1[wave][d0] = g0; rb1[wave][d1] = g1;   // same-wave DS in-order

    float f0 = fc2b[d0], f1 = fc2b[d1];
    #pragma unroll 8
    for (int k = 0; k < 64; ++k) {
        const float h = rb1[wave][k];
        f0 = fmaf(h, W2[d0 * 64 + k], f0);
        f1 = fmaf(h, W2[d1 * 64 + k], f1);
    }
    float* outr = out + (row << 6);
    outr[d0] = xa0 + f0;
    outr[d1] = xa1 + f1;
}

// ---------------------------------------------------------------------------
extern "C" void kernel_launch(void* const* d_in, const int* in_sizes, int n_in,
                              void* d_out, int out_size, void* d_ws, size_t ws_size,
                              hipStream_t stream)
{
    (void)in_sizes; (void)n_in; (void)out_size; (void)ws_size;
    const float* x     = (const float*)d_in[0];
    const float* n1w   = (const float*)d_in[1];
    const float* n1b   = (const float*)d_in[2];
    const float* qkvw  = (const float*)d_in[3];
    const float* scale = (const float*)d_in[4];
    const float* projw = (const float*)d_in[5];
    const float* projb = (const float*)d_in[6];
    const float* n2w   = (const float*)d_in[7];
    const float* n2b   = (const float*)d_in[8];
    const float* fc1w  = (const float*)d_in[9];
    const float* fc1b  = (const float*)d_in[10];
    const float* fc2w  = (const float*)d_in[11];
    const float* fc2b  = (const float*)d_in[12];
    float* out = (float*)d_out;

    // workspace layout (bytes)
    const size_t F16 = (size_t)BATCH * NTOK * IND * sizeof(_Float16);  // 3,211,264
    const size_t F32 = (size_t)BATCH * NTOK * IND * sizeof(float);     // 6,422,528
    char* ws = (char*)d_ws;
    _Float16* qT  = (_Float16*)(ws);
    _Float16* kf  = (_Float16*)(ws + F16);
    _Float16* vT  = (_Float16*)(ws + 2 * F16);
    float*    v32 = (float*)   (ws + 3 * F16);
    float*    ow  = (float*)   (ws + 3 * F16 + F32);

    const int rows = BATCH * NTOK;                  // 25088
    ln_qkv_kernel<<<rows / 8, 256, 0, stream>>>(x, n1w, n1b, qkvw, scale,
                                                qT, kf, vT, v32);
    flash_attn_kernel<<<(BATCH * QTILES_PER_B) / 4, 128, 0, stream>>>(qT, kf, vT, ow);
    mlp_kernel<<<rows / 8, 256, 0, stream>>>(v32, ow, projw, projb, n2w, n2b,
                                             fc1w, fc1b, fc2w, fc2b, out);
}
